// NodeUpdateLayer_16020228014616
// MI455X (gfx1250) — matrix operand order repaired and checked by reading, not compiled, not run on hardware
//
#include <hip/hip_runtime.h>
#include <hip/hip_bf16.h>
#include <stdint.h>

#define NN 50000
#define NE 800000
#define DD 128
#define EF 64
#define NHEAD 8
#define KVIN 320
#define EPSV 1e-5f

typedef __attribute__((ext_vector_type(16))) _Float16 v16h;
typedef __attribute__((ext_vector_type(8)))  _Float16 half8;
typedef __attribute__((ext_vector_type(8)))  float    v8f;

union V16HU { v16h v; half8 h[2]; };

// ---- WMMA fragment loaders -------------------------------------------------
// A (16x32 f16): lane m=lane&15; lane<16 -> K {kb..kb+7, kb+16..kb+23}
//                lane>=16 -> K {kb+8..kb+15, kb+24..kb+31}
__device__ inline v16h load_a_frag(const _Float16* x, int pitch, int kb, int lane) {
  int m = lane & 15;
  int koff = (lane < 16) ? 0 : 8;
  const _Float16* p = x + m * pitch + kb + koff;
  V16HU u;
  u.h[0] = *(const half8*)(p);
  u.h[1] = *(const half8*)(p + 16);
  return u.v;
}

// B (32x16 f16) from f16 W^T stored as [Ncols][K]: lane n=col; lane<16 -> K {kb..kb+7, kb+16..kb+23}
// lane>=16 -> K {kb+8..kb+15, kb+24..kb+31}.
__device__ inline v16h load_b_frag(const _Float16* wt, int K, int tile, int kb, int lane) {
  int n = tile * 16 + (lane & 15);
  int koff = (lane < 16) ? 0 : 8;
  const _Float16* p = wt + (size_t)n * K + kb + koff;
  V16HU u;
  u.h[0] = *(const half8*)(p);
  u.h[1] = *(const half8*)(p + 16);
  return u.v;
}

// ---- Fused MLP core: GEMM(DINx128) -> LayerNorm -> ReLU -> GEMM(128x128) ---
// One wave computes 16 rows x 128 cols. Hidden staged through per-wave LDS
// for the cross-lane LayerNorm reduction.
template<int DIN>
__device__ void mlp_core(const _Float16* xw,      // this wave's 16 rows, pitch DIN+8
                         float* hidf,             // per-wave [16][132] f32
                         _Float16* hid16,         // per-wave [16][136] f16
                         float* mrs,              // per-wave [16][2] (mu, rsig)
                         const _Float16* w1t, const float* b1,
                         const float* g, const float* be,
                         const _Float16* w2t, const float* b2,
                         int lane, v8f out[8]) {
  constexpr int XP = DIN + 8;
  v8f acc[8];
#pragma unroll
  for (int t = 0; t < 8; ++t) {
    float bv = b1[t * 16 + (lane & 15)];
#pragma unroll
    for (int j = 0; j < 8; ++j) acc[t][j] = bv;
  }
  for (int kb = 0; kb < DIN; kb += 32) {
    v16h a = load_a_frag(xw, XP, kb, lane);
#pragma unroll
    for (int t = 0; t < 8; ++t) {
      v16h b = load_b_frag(w1t, DIN, t, kb, lane);
      acc[t] = __builtin_amdgcn_wmma_f32_16x16x32_f16(false, a, false, b,
                                                      (short)0, acc[t], false, false);
    }
  }
  // stage hidden (C layout: vgpr j -> row j (lanes 0-15) / row j+8 (lanes 16-31))
  {
    int col = lane & 15;
    int rbase = (lane < 16) ? 0 : 8;
#pragma unroll
    for (int t = 0; t < 8; ++t)
#pragma unroll
      for (int j = 0; j < 8; ++j)
        hidf[(rbase + j) * 132 + t * 16 + col] = acc[t][j];
  }
  __syncthreads();
  // LayerNorm stats: lane (r, half) sums 64 values, combine halves via shuffle
  {
    int row = lane & 15;
    int half = lane >> 4;
    float s = 0.f, s2 = 0.f;
    for (int c = half * 64; c < half * 64 + 64; ++c) {
      float x = hidf[row * 132 + c];
      s += x; s2 += x * x;
    }
    s  += __shfl_xor(s, 16, 32);
    s2 += __shfl_xor(s2, 16, 32);
    if (half == 0) {
      float mu  = s * (1.f / 128.f);
      float var = s2 * (1.f / 128.f) - mu * mu;
      mrs[row * 2 + 0] = mu;
      mrs[row * 2 + 1] = rsqrtf(var + EPSV);
    }
  }
  __syncthreads();
  // apply LN + ReLU, convert to f16
  for (int idx = lane; idx < 16 * 128; idx += 32) {
    int row = idx >> 7, c = idx & 127;
    float x = hidf[row * 132 + c];
    float y = (x - mrs[row * 2]) * mrs[row * 2 + 1] * g[c] + be[c];
    y = fmaxf(y, 0.f);
    hid16[row * 136 + c] = (_Float16)y;
  }
  __syncthreads();
  // GEMM2: 128 -> 128
#pragma unroll
  for (int t = 0; t < 8; ++t) {
    float bv = b2[t * 16 + (lane & 15)];
#pragma unroll
    for (int j = 0; j < 8; ++j) out[t][j] = bv;
  }
  for (int kb = 0; kb < 128; kb += 32) {
    v16h a = load_a_frag(hid16, 136, kb, lane);
#pragma unroll
    for (int t = 0; t < 8; ++t) {
      v16h b = load_b_frag(w2t, 128, t, kb, lane);
      out[t] = __builtin_amdgcn_wmma_f32_16x16x32_f16(false, a, false, b,
                                                      (short)0, out[t], false, false);
    }
  }
  __syncthreads();
}

// ---- kernels ---------------------------------------------------------------
__global__ void init_zero_kernel(unsigned int* nodemax, float* den, float* outacc) {
  int gid = blockIdx.x * blockDim.x + threadIdx.x;
  if (gid < NN * NHEAD) { nodemax[gid] = 0u; den[gid] = 0.f; }
  if (gid < NN * DD) outacc[gid] = 0.f;
}

__global__ void transpose_w_kernel(const float* __restrict__ w, _Float16* __restrict__ wt, int K) {
  int gid = blockIdx.x * blockDim.x + threadIdx.x;
  if (gid >= 128 * K) return;
  int n = gid / K, k = gid - n * K;
  wt[gid] = (_Float16)w[(size_t)k * 128 + n];
}

__global__ __launch_bounds__(64) void q_mlp_kernel(
    const float* __restrict__ h,
    const _Float16* __restrict__ w1t, const float* __restrict__ b1,
    const float* __restrict__ g, const float* __restrict__ be,
    const _Float16* __restrict__ w2t, const float* __restrict__ b2,
    float* __restrict__ qout) {
  __shared__ _Float16 xs[32 * 136];
  __shared__ float hidf[2][16 * 132];
  __shared__ _Float16 hid16[2][16 * 136];
  __shared__ float mrs[2][32];
  int tid = threadIdx.x, wave = tid >> 5, lane = tid & 31;
  int base = blockIdx.x * 32;
  for (int idx = tid; idx < 32 * 128; idx += 64) {
    int r = idx >> 7, c = idx & 127;
    int node = base + r;
    float xv = (node < NN) ? h[(size_t)node * 128 + c] : 0.f;
    xs[r * 136 + c] = (_Float16)xv;
  }
  __syncthreads();
  v8f out[8];
  mlp_core<128>(xs + wave * 16 * 136, hidf[wave], hid16[wave], mrs[wave],
                w1t, b1, g, be, w2t, b2, lane, out);
  int col = lane & 15, rbase = (lane < 16) ? 0 : 8;
#pragma unroll
  for (int t = 0; t < 8; ++t)
#pragma unroll
    for (int j = 0; j < 8; ++j) {
      int node = base + wave * 16 + rbase + j;
      if (node < NN) qout[(size_t)node * 128 + t * 16 + col] = out[t][j];
    }
}

__global__ __launch_bounds__(64) void edge_kernel(
    const float* __restrict__ h, const float* __restrict__ edge_feat,
    const float* __restrict__ e_w, const int* __restrict__ eidx,
    const float* __restrict__ q,
    const _Float16* kw1t, const float* kb1, const float* kg, const float* kbe,
    const _Float16* kw2t, const float* kb2,
    const _Float16* vw1t, const float* vb1, const float* vg, const float* vbe,
    const _Float16* vw2t, const float* vb2,
    float* __restrict__ logits, _Float16* __restrict__ vbuf,
    unsigned int* __restrict__ nodemax) {
  __shared__ _Float16 xs[32 * (KVIN + 8)];
  __shared__ float hidf[2][16 * 132];
  __shared__ _Float16 hid16[2][16 * 136];
  __shared__ float mrs[2][32];
  __shared__ int sDst[32], sSrc[32];
  __shared__ float sEw[32];
  int tid = threadIdx.x, wave = tid >> 5, lane = tid & 31;
  int eBase = blockIdx.x * 32;
  if (tid < 32) {
    int e = eBase + tid;
    sSrc[tid] = eidx[e];
    sDst[tid] = eidx[NE + e];
    sEw[tid]  = e_w[e];
  }
  __syncthreads();
  // build kv = [edge_feat | h[dst] | h[src]] as f16
  for (int idx = tid; idx < 32 * KVIN; idx += 64) {
    int r = idx / KVIN, c = idx - r * KVIN;
    int e = eBase + r;
    float xv;
    if (c < 64)       xv = edge_feat[(size_t)e * 64 + c];
    else if (c < 192) xv = h[(size_t)sDst[r] * 128 + (c - 64)];
    else              xv = h[(size_t)sSrc[r] * 128 + (c - 192)];
    xs[r * (KVIN + 8) + c] = (_Float16)xv;
  }
  __syncthreads();
  v8f out[8];
  // ---- K MLP ----
  mlp_core<KVIN>(xs + wave * 16 * (KVIN + 8), hidf[wave], hid16[wave], mrs[wave],
                 kw1t, kb1, kg, kbe, kw2t, kb2, lane, out);
  // stage k rows for per-head dot with q[dst]
  {
    int col = lane & 15, rbase = (lane < 16) ? 0 : 8;
#pragma unroll
    for (int t = 0; t < 8; ++t)
#pragma unroll
      for (int j = 0; j < 8; ++j)
        hidf[wave][(rbase + j) * 132 + t * 16 + col] = out[t][j];
  }
  __syncthreads();
  // logits + scatter-max (monotone uint key)
  {
    int eloc = lane & 15;
    int e = eBase + wave * 16 + eloc;
    int dst = sDst[wave * 16 + eloc];
#pragma unroll
    for (int p = 0; p < 4; ++p) {
      int head = (lane >> 4) * 4 + p;
      float d = 0.f;
#pragma unroll
      for (int j = 0; j < 16; ++j)
        d += hidf[wave][eloc * 132 + head * 16 + j] * q[(size_t)dst * 128 + head * 16 + j];
      d *= 0.25f;   // 1/sqrt(16)
      logits[(size_t)e * 8 + head] = d;
      unsigned int b = __float_as_uint(d);
      unsigned int key = (b & 0x80000000u) ? ~b : (b | 0x80000000u);
      atomicMax(&nodemax[dst * 8 + head], key);
    }
  }
  __syncthreads();
  // ---- V MLP ----
  mlp_core<KVIN>(xs + wave * 16 * (KVIN + 8), hidf[wave], hid16[wave], mrs[wave],
                 vw1t, vb1, vg, vbe, vw2t, vb2, lane, out);
  {
    int col = lane & 15, rbase = (lane < 16) ? 0 : 8;
#pragma unroll
    for (int t = 0; t < 8; ++t)
#pragma unroll
      for (int j = 0; j < 8; ++j) {
        int r = wave * 16 + rbase + j;
        int e = eBase + r;
        vbuf[(size_t)e * 128 + t * 16 + col] = (_Float16)(out[t][j] * sEw[r]);
      }
  }
}

__global__ void softmax_pass2(const int* __restrict__ eidx, float* __restrict__ logits,
                              const unsigned int* __restrict__ nodemax, float* __restrict__ den) {
  int gid = blockIdx.x * blockDim.x + threadIdx.x;
  if (gid >= NE * NHEAD) return;
  int e = gid >> 3, head = gid & 7;
  int dst = eidx[NE + e];
  unsigned int key = nodemax[dst * 8 + head];
  unsigned int b = (key & 0x80000000u) ? (key ^ 0x80000000u) : ~key;
  float mx = __uint_as_float(b);
  float ex = __expf(logits[gid] - mx);
  logits[gid] = ex;
  atomicAdd(&den[dst * 8 + head], ex);
}

__global__ void scatter_pass3(const int* __restrict__ eidx, const float* __restrict__ logits,
                              const _Float16* __restrict__ vbuf, float* __restrict__ outacc) {
  int gid = blockIdx.x * blockDim.x + threadIdx.x;
  if (gid >= NE * 4) return;
  int e = gid >> 2, qn = gid & 3;
  int dst = eidx[NE + e];
  float a0 = logits[(size_t)e * 8 + qn * 2];
  float a1 = logits[(size_t)e * 8 + qn * 2 + 1];
  const _Float16* vp = vbuf + (size_t)e * 128 + qn * 32;
  float* op = outacc + (size_t)dst * 128 + qn * 32;
#pragma unroll
  for (int c = 0; c < 32; ++c) {
    float al = (c < 16) ? a0 : a1;
    atomicAdd(&op[c], al * (float)vp[c]);
  }
}

__global__ __launch_bounds__(64) void out_mlp_kernel(
    const float* __restrict__ h, const float* __restrict__ outacc,
    const float* __restrict__ den,
    const _Float16* w1t, const float* b1, const float* g, const float* be,
    const _Float16* w2t, const float* b2, float* __restrict__ outp) {
  __shared__ _Float16 xs[32 * 264];
  __shared__ float hidf[2][16 * 132];
  __shared__ _Float16 hid16[2][16 * 136];
  __shared__ float mrs[2][32];
  int tid = threadIdx.x, wave = tid >> 5, lane = tid & 31;
  int base = blockIdx.x * 32;
  for (int idx = tid; idx < 32 * 256; idx += 64) {
    int r = idx >> 8, c = idx & 255;
    int node = base + r;
    float xv = 0.f;
    if (node < NN) {
      if (c < 128) {
        float dd = den[node * 8 + (c >> 4)];
        xv = (dd > 0.f) ? outacc[(size_t)node * 128 + c] / dd : 0.f;
      } else {
        xv = h[(size_t)node * 128 + (c - 128)];
      }
    }
    xs[r * 264 + c] = (_Float16)xv;
  }
  __syncthreads();
  v8f out[8];
  mlp_core<256>(xs + wave * 16 * 264, hidf[wave], hid16[wave], mrs[wave],
                w1t, b1, g, be, w2t, b2, lane, out);
  int col = lane & 15, rbase = (lane < 16) ? 0 : 8;
#pragma unroll
  for (int t = 0; t < 8; ++t)
#pragma unroll
    for (int j = 0; j < 8; ++j) {
      int node = base + wave * 16 + rbase + j;
      if (node < NN) outp[(size_t)node * 128 + t * 16 + col] = out[t][j];
    }
}

// ---- host launcher ---------------------------------------------------------
extern "C" void kernel_launch(void* const* d_in, const int* in_sizes, int n_in,
                              void* d_out, int out_size, void* d_ws, size_t ws_size,
                              hipStream_t stream) {
  const float* h         = (const float*)d_in[0];
  const float* edge_feat = (const float*)d_in[1];
  const float* e_w       = (const float*)d_in[2];
  const int*   eidx      = (const int*)d_in[3];
  const float* W[8];  const float* B1[4]; const float* G[4]; const float* BE[4]; const float* B2[4];
  // order: hk(4..9), hv(10..15), hq(16..21), no(22..27): W1,b1,g,be,W2,b2
  const float* W1s[4] = {(const float*)d_in[4],  (const float*)d_in[10],
                         (const float*)d_in[16], (const float*)d_in[22]};
  const float* b1s[4] = {(const float*)d_in[5],  (const float*)d_in[11],
                         (const float*)d_in[17], (const float*)d_in[23]};
  const float* gs[4]  = {(const float*)d_in[6],  (const float*)d_in[12],
                         (const float*)d_in[18], (const float*)d_in[24]};
  const float* bes[4] = {(const float*)d_in[7],  (const float*)d_in[13],
                         (const float*)d_in[19], (const float*)d_in[25]};
  const float* W2s[4] = {(const float*)d_in[8],  (const float*)d_in[14],
                         (const float*)d_in[20], (const float*)d_in[26]};
  const float* b2s[4] = {(const float*)d_in[9],  (const float*)d_in[15],
                         (const float*)d_in[21], (const float*)d_in[27]};
  (void)W; (void)B1; (void)G; (void)BE; (void)B2;
  (void)in_sizes; (void)n_in; (void)out_size; (void)ws_size;

  char* ws = (char*)d_ws;
  size_t off = 0;
  auto alloc = [&](size_t bytes) -> void* {
    void* p = ws + off;
    off += (bytes + 255) & ~(size_t)255;
    return p;
  };
  float*        qbuf    = (float*)alloc((size_t)NN * 128 * 4);
  _Float16*     vbuf    = (_Float16*)alloc((size_t)NE * 128 * 2);
  float*        logits  = (float*)alloc((size_t)NE * 8 * 4);
  unsigned int* nodemax = (unsigned int*)alloc((size_t)NN * 8 * 4);
  float*        den     = (float*)alloc((size_t)NN * 8 * 4);
  float*        outacc  = (float*)alloc((size_t)NN * 128 * 4);
  const int Kdims[4] = {KVIN, KVIN, 128, 256};  // W1 K for hk, hv, hq, no
  _Float16* W1T[4]; _Float16* W2T[4];
  for (int i = 0; i < 4; ++i) {
    W1T[i] = (_Float16*)alloc((size_t)128 * Kdims[i] * 2);
    W2T[i] = (_Float16*)alloc((size_t)128 * 128 * 2);
  }

  // 1) init accumulators
  init_zero_kernel<<<(NN * 128 + 255) / 256, 256, 0, stream>>>(nodemax, den, outacc);
  // 2) weight transpose + f16 convert
  for (int i = 0; i < 4; ++i) {
    transpose_w_kernel<<<(128 * Kdims[i] + 255) / 256, 256, 0, stream>>>(W1s[i], W1T[i], Kdims[i]);
    transpose_w_kernel<<<(128 * 128 + 255) / 256, 256, 0, stream>>>(W2s[i], W2T[i], 128);
  }
  // 3) q = MLP_q(h)
  q_mlp_kernel<<<(NN + 31) / 32, 64, 0, stream>>>(h, W1T[2], b1s[2], gs[2], bes[2],
                                                  W2T[2], b2s[2], qbuf);
  // 4) per-edge k/v MLPs + logits + scatter-max
  edge_kernel<<<NE / 32, 64, 0, stream>>>(h, edge_feat, e_w, eidx, qbuf,
                                          W1T[0], b1s[0], gs[0], bes[0], W2T[0], b2s[0],
                                          W1T[1], b1s[1], gs[1], bes[1], W2T[1], b2s[1],
                                          logits, vbuf, nodemax);
  // 5) exp(logit - max), denominator
  softmax_pass2<<<(NE * 8 + 255) / 256, 256, 0, stream>>>(eidx, logits, nodemax, den);
  // 6) scatter ex*v into outacc
  scatter_pass3<<<(NE * 4 + 255) / 256, 256, 0, stream>>>(eidx, logits, vbuf, outacc);
  // 7) final MLP on [outacc/den | h]
  out_mlp_kernel<<<(NN + 31) / 32, 64, 0, stream>>>(h, outacc, den,
                                                    W1T[3], b1s[3], gs[3], bes[3],
                                                    W2T[3], b2s[3], (float*)d_out);
}
